// ConceptBottleneck_51934744543354
// MI455X (gfx1250) — matrix operand order repaired and checked by reading, not compiled, not run on hardware
//
#include <hip/hip_runtime.h>
#include <hip/hip_bf16.h>

// Concept Bottleneck on MI455X (gfx1250): all einsums as v_wmma_f32_16x16x32_f16.
// Phase 0: convert x -> f16, repack all weights into WMMA B-fragment order.
// Phase 1: bank_kernel (pos, neg): fused 768->64->64->16 MLP, M=32 rows/wave,
//          W1 streamed through a double-buffered LDS ring (async-to-LDS when
//          available), W2/W3 LDS-resident.
// Phase 2: cp_gate_kernel: fused 32->64->64->1 MLP + gate + blend, weights in LDS.

typedef __attribute__((ext_vector_type(16))) _Float16 v16h;
typedef __attribute__((ext_vector_type(8)))  _Float16 v8h;
typedef __attribute__((ext_vector_type(8)))  float    v8f;
typedef int v4i __attribute__((vector_size(16)));

#define BB 8192
#define INDIM 768
#define NC 64
#define HD 64
#define ED 16

// ---- CDNA5 async global->LDS path (guarded; falls back to reg bounce) ----
#if defined(__has_builtin)
#if __has_builtin(__builtin_amdgcn_global_load_async_to_lds_b128)
#define ASYNC_LDS 1
#endif
#endif
#ifndef ASYNC_LDS
#define ASYNC_LDS 0
#endif

#if ASYNC_LDS
static __device__ __forceinline__ void async_cp16(const _Float16* g, _Float16* l) {
  __builtin_amdgcn_global_load_async_to_lds_b128(
      (v4i*)g, (v4i*)l, 0, 0);
}
static __device__ __forceinline__ void async_wait_all() {
#if defined(__has_builtin) && __has_builtin(__builtin_amdgcn_s_wait_asynccnt)
  __builtin_amdgcn_s_wait_asynccnt(0);
#else
  asm volatile("s_wait_asynccnt 0" ::: "memory");
#endif
}
#endif

static __device__ __forceinline__ v8f wmma16(v16h a, v16h b, v8f c) {
  return __builtin_amdgcn_wmma_f32_16x16x32_f16(false, a, false, b, (short)0, c,
                                                false, false);
}

static __device__ __forceinline__ v16h frag_from(const _Float16* p0,
                                                 const _Float16* p1) {
  v8h lo = *reinterpret_cast<const v8h*>(p0);
  v8h hi = *reinterpret_cast<const v8h*>(p1);
  v16h r;
#pragma unroll
  for (int i = 0; i < 8; ++i) { r[i] = lo[i]; r[i + 8] = hi[i]; }
  return r;
}

static __device__ __forceinline__ v8f vzero8() {
  v8f z;
#pragma unroll
  for (int i = 0; i < 8; ++i) z[i] = 0.f;
  return z;
}

// ---------- Phase 0: conversion / repack ----------

__global__ void cvt_f16_kernel(const float* __restrict__ src,
                               _Float16* __restrict__ dst, int n) {
  int i = blockIdx.x * 256 + threadIdx.x;
  if (i < n) dst[i] = (_Float16)src[i];
}

// Repack W[C][Ktot][Nsrc] (f32 row-major) into per-concept WMMA B-fragment
// blocks of 512 f16: blk=(kt*(Ndst/16)+nt); lane l holds 16 contiguous f16 at
// blk*512 + l*16; element j <-> K = kt*32 + (l/16)*8 + j + (j>=8 ? 8 : 0), N = nt*16 + l%16.
__global__ void repack_w_kernel(const float* __restrict__ W,
                                _Float16* __restrict__ out, int Ktot, int Nsrc,
                                int Ndst, int total) {
  int t = blockIdx.x * 256 + threadIdx.x;
  if (t >= total) return;
  int S = Ktot * Ndst;
  int c = t / S, rem = t % S;
  int blk = rem >> 9, wit = rem & 511;
  int ln = wit >> 4, j = wit & 15;
  int nBlocks = Ndst >> 4;
  int kt = blk / nBlocks, nt = blk % nBlocks;
  int K = kt * 32 + (ln >> 4) * 8 + j + (j >= 8 ? 8 : 0);
  int N = nt * 16 + (ln & 15);
  float v = (N < Nsrc) ? W[((size_t)c * Ktot + K) * Nsrc + N] : 0.f;
  out[t] = (_Float16)v;
}

// ---------- Phase 1: pos/neg bank (768 -> 64 -> 64 -> 16, fused) ----------
// 256 threads = 8 waves; wave owns M=32 rows; WG covers 256 rows of one concept.
// W1 streamed: 6 chunks of 4 k-steps (16KB), double buffered in LDS.

#define KT_CHUNK 4
#define N_CHUNKS (INDIM / 32 / KT_CHUNK)          // 6
#define CHUNK_H  (KT_CHUNK * 4 * 512)             // 8192 f16 per chunk

__global__ __launch_bounds__(256) void bank_kernel(
    const _Float16* __restrict__ xh,   // [B, IN] f16 row-major
    const _Float16* __restrict__ w1,   // [C] fragment-ordered, Ktot=768 Ndst=64
    const float* __restrict__ b1,      // [C, H]
    const _Float16* __restrict__ w2,   // [C] Ktot=64 Ndst=64
    const float* __restrict__ b2,      // [C, H]
    const _Float16* __restrict__ w3,   // [C] Ktot=64 Ndst=16
    const float* __restrict__ b3,      // [C, E]
    float* __restrict__ out)           // [B, C, E] f32
{
  // 0..32KB : W1 double buffer (2 x 8192 f16), later overlaid by hbuf
  //           hbuf: 8 waves x 32 rows x 64 f16  (32KB)
  // 32..40KB: W2 (4096 f16)   40..42KB: W3 (1024 f16)
  __shared__ __align__(16) char smem[43008];
  _Float16* w1buf = (_Float16*)smem;
  _Float16* hbuf  = (_Float16*)smem;
  _Float16* w2buf = (_Float16*)(smem + 32768);
  _Float16* w3buf = (_Float16*)(smem + 40960);

  const int tid = threadIdx.x;
  const int wv = tid >> 5, lane = tid & 31;
  const int g = lane >> 4, ln = lane & 15;
  const int c = blockIdx.y;
  const int mb = blockIdx.x * 256 + wv * 32;

  const _Float16* w1c = w1 + (size_t)c * INDIM * HD;
  const _Float16* w2c = w2 + (size_t)c * HD * HD;
  const _Float16* w3c = w3 + (size_t)c * HD * ED;

  // ---- stage W2/W3 (once) and W1 chunk 0 ----
#pragma unroll
  for (int i = 0; i < 2; ++i)
    *reinterpret_cast<v8h*>(w2buf + tid * 8 + i * 2048) =
        *reinterpret_cast<const v8h*>(w2c + tid * 8 + i * 2048);
  if (tid < 128)
    *reinterpret_cast<v8h*>(w3buf + tid * 8) =
        *reinterpret_cast<const v8h*>(w3c + tid * 8);
#if ASYNC_LDS
#pragma unroll
  for (int i = 0; i < 4; ++i)
    async_cp16(w1c + tid * 8 + i * 2048, w1buf + tid * 8 + i * 2048);
  async_wait_all();
#else
  {
    v8h t0[4];
#pragma unroll
    for (int i = 0; i < 4; ++i)
      t0[i] = *reinterpret_cast<const v8h*>(w1c + tid * 8 + i * 2048);
#pragma unroll
    for (int i = 0; i < 4; ++i)
      *reinterpret_cast<v8h*>(w1buf + tid * 8 + i * 2048) = t0[i];
  }
#endif
  __syncthreads();

  // ---- layer 1: M=32 N=64 K=768, W1 through LDS ring ----
  v8f acc[2][4];
#pragma unroll
  for (int h = 0; h < 2; ++h)
#pragma unroll
    for (int nt = 0; nt < 4; ++nt) acc[h][nt] = vzero8();

  const _Float16* xrow0 = xh + (size_t)(mb + ln) * INDIM;
  const _Float16* xrow1 = xh + (size_t)(mb + 16 + ln) * INDIM;

  for (int ch = 0; ch < N_CHUNKS; ++ch) {
    const int cur = ch & 1;
    const bool haveNext = (ch + 1) < N_CHUNKS;
#if !ASYNC_LDS
    v8h tn[4];
#endif
    if (haveNext) {
      const _Float16* src = w1c + (size_t)(ch + 1) * CHUNK_H;
#if ASYNC_LDS
      _Float16* dst = w1buf + (1 - cur) * CHUNK_H;
#pragma unroll
      for (int i = 0; i < 4; ++i)
        async_cp16(src + tid * 8 + i * 2048, dst + tid * 8 + i * 2048);
#else
#pragma unroll
      for (int i = 0; i < 4; ++i)
        tn[i] = *reinterpret_cast<const v8h*>(src + tid * 8 + i * 2048);
#endif
    }

    const _Float16* wb = w1buf + cur * CHUNK_H;
#pragma unroll
    for (int k4 = 0; k4 < KT_CHUNK; ++k4) {
      const int kb = (ch * KT_CHUNK + k4) * 32 + g * 8;
      v16h a0 = frag_from(xrow0 + kb, xrow0 + kb + 16);
      v16h a1 = frag_from(xrow1 + kb, xrow1 + kb + 16);
#pragma unroll
      for (int nt = 0; nt < 4; ++nt) {
        const _Float16* bp = wb + ((k4 * 4 + nt) << 9) + lane * 16;
        v16h b = frag_from(bp, bp + 8);
        acc[0][nt] = wmma16(a0, b, acc[0][nt]);
        acc[1][nt] = wmma16(a1, b, acc[1][nt]);
      }
    }

    if (haveNext) {
#if ASYNC_LDS
      async_wait_all();
#else
      _Float16* dst = w1buf + (1 - cur) * CHUNK_H;
#pragma unroll
      for (int i = 0; i < 4; ++i)
        *reinterpret_cast<v8h*>(dst + tid * 8 + i * 2048) = tn[i];
#endif
    }
    __syncthreads();
  }

  // ---- bias + relu -> hbuf (overlays W1 ring; all waves are past it) ----
  _Float16* hw = hbuf + wv * 2048;  // 32 rows x 64
#pragma unroll
  for (int h = 0; h < 2; ++h)
#pragma unroll
    for (int nt = 0; nt < 4; ++nt) {
      float bias = b1[c * HD + nt * 16 + ln];
#pragma unroll
      for (int r = 0; r < 8; ++r) {
        float v = acc[h][nt][r] + bias;
        v = v > 0.f ? v : 0.f;
        hw[(h * 16 + r + 8 * g) * HD + nt * 16 + ln] = (_Float16)v;
      }
    }

  // ---- layer 2: M=32 N=64 K=64 (A from hbuf, B from w2buf) ----
  v8f acc2[2][4];
#pragma unroll
  for (int h = 0; h < 2; ++h)
#pragma unroll
    for (int nt = 0; nt < 4; ++nt) acc2[h][nt] = vzero8();
#pragma unroll
  for (int kt = 0; kt < 2; ++kt) {
    const int kb = kt * 32 + g * 8;
    v16h a0 = frag_from(&hw[ln * HD + kb], &hw[ln * HD + kb + 16]);
    v16h a1 = frag_from(&hw[(16 + ln) * HD + kb], &hw[(16 + ln) * HD + kb + 16]);
#pragma unroll
    for (int nt = 0; nt < 4; ++nt) {
      const _Float16* bp = w2buf + ((kt * 4 + nt) << 9) + lane * 16;
      v16h b = frag_from(bp, bp + 8);
      acc2[0][nt] = wmma16(a0, b, acc2[0][nt]);
      acc2[1][nt] = wmma16(a1, b, acc2[1][nt]);
    }
  }
#pragma unroll
  for (int h = 0; h < 2; ++h)
#pragma unroll
    for (int nt = 0; nt < 4; ++nt) {
      float bias = b2[c * HD + nt * 16 + ln];
#pragma unroll
      for (int r = 0; r < 8; ++r) {
        float v = acc2[h][nt][r] + bias;
        v = v > 0.f ? v : 0.f;
        hw[(h * 16 + r + 8 * g) * HD + nt * 16 + ln] = (_Float16)v;
      }
    }

  // ---- layer 3: M=32 N=16 K=64 ----
  v8f acc3[2];
  acc3[0] = vzero8();
  acc3[1] = vzero8();
#pragma unroll
  for (int kt = 0; kt < 2; ++kt) {
    const int kb = kt * 32 + g * 8;
    v16h a0 = frag_from(&hw[ln * HD + kb], &hw[ln * HD + kb + 16]);
    v16h a1 = frag_from(&hw[(16 + ln) * HD + kb], &hw[(16 + ln) * HD + kb + 16]);
    const _Float16* bp = w3buf + ((size_t)kt << 9) + lane * 16;
    v16h b = frag_from(bp, bp + 8);
    acc3[0] = wmma16(a0, b, acc3[0]);
    acc3[1] = wmma16(a1, b, acc3[1]);
  }
  const float b3v = b3[c * ED + ln];
#pragma unroll
  for (int h = 0; h < 2; ++h)
#pragma unroll
    for (int r = 0; r < 8; ++r) {
      int m = h * 16 + r + 8 * g;
      out[((size_t)(mb + m) * NC + c) * ED + ln] = acc3[h][r] + b3v;
    }
}

// ---------- Phase 2: cp bank (32 -> 64 -> 64 -> 1) + gate + blend ----------

__global__ __launch_bounds__(256) void cp_gate_kernel(
    const float* __restrict__ pos,   // [B, C, E]
    const float* __restrict__ neg,   // [B, C, E]
    const _Float16* __restrict__ w1, // [C] Ktot=32 Ndst=64
    const float* __restrict__ b1,    // [C, H]
    const _Float16* __restrict__ w2, // [C] Ktot=64 Ndst=64
    const float* __restrict__ b2,    // [C, H]
    const _Float16* __restrict__ w3, // [C] Ktot=64 Ndst=16 (N=1 zero-padded)
    const float* __restrict__ b3,    // [C, 1]
    float* __restrict__ emb,         // [B, E*C]  (emb[b][e*C + c])
    float* __restrict__ con)         // [B, C]
{
  __shared__ _Float16 xcb[8][16 * 32];
  __shared__ _Float16 hbuf[8][16 * HD];
  __shared__ _Float16 cw1buf[32 * HD];   // 4KB
  __shared__ _Float16 cw2buf[HD * HD];   // 8KB
  __shared__ _Float16 cw3buf[HD * ED];   // 2KB
  __shared__ float cbf[8][16];

  const int tid = threadIdx.x;
  const int wv = tid >> 5, lane = tid & 31;
  const int g = lane >> 4, ln = lane & 15;
  const int c = blockIdx.y;
  const int mb = blockIdx.x * 128 + wv * 16;

  // stage cp weights for this concept
  *reinterpret_cast<v8h*>(cw1buf + tid * 8) =
      *reinterpret_cast<const v8h*>(w1 + (size_t)c * 32 * HD + tid * 8);
#pragma unroll
  for (int i = 0; i < 2; ++i)
    *reinterpret_cast<v8h*>(cw2buf + tid * 8 + i * 2048) =
        *reinterpret_cast<const v8h*>(w2 + (size_t)c * HD * HD + tid * 8 + i * 2048);
  if (tid < 128)
    *reinterpret_cast<v8h*>(cw3buf + tid * 8) =
        *reinterpret_cast<const v8h*>(w3 + (size_t)c * HD * ED + tid * 8);

  // build xc: g==0 lanes pack pos row, g==1 lanes pack neg row (f32 -> f16)
  {
    const float* srow = (g == 0 ? pos : neg) + ((size_t)(mb + ln) * NC + c) * ED;
    v8h t0, t1;
#pragma unroll
    for (int i = 0; i < 8; ++i) { t0[i] = (_Float16)srow[i]; t1[i] = (_Float16)srow[8 + i]; }
    *reinterpret_cast<v8h*>(&xcb[wv][ln * 32 + g * 16]) = t0;
    *reinterpret_cast<v8h*>(&xcb[wv][ln * 32 + g * 16 + 8]) = t1;
  }
  __syncthreads();

  // ---- layer 1: M=16 N=64 K=32 ----
  v8f acc[4];
#pragma unroll
  for (int nt = 0; nt < 4; ++nt) acc[nt] = vzero8();
  {
    v16h a = frag_from(&xcb[wv][ln * 32 + g * 8], &xcb[wv][ln * 32 + 16 + g * 8]);
#pragma unroll
    for (int nt = 0; nt < 4; ++nt) {
      const _Float16* bp = cw1buf + (nt << 9) + lane * 16;
      v16h b = frag_from(bp, bp + 8);
      acc[nt] = wmma16(a, b, acc[nt]);
    }
  }
#pragma unroll
  for (int nt = 0; nt < 4; ++nt) {
    float bias = b1[c * HD + nt * 16 + ln];
#pragma unroll
    for (int r = 0; r < 8; ++r) {
      float v = acc[nt][r] + bias;
      v = v > 0.f ? v : 0.f;
      hbuf[wv][(r + 8 * g) * HD + nt * 16 + ln] = (_Float16)v;
    }
  }

  // ---- layer 2: M=16 N=64 K=64 ----
  v8f acc2[4];
#pragma unroll
  for (int nt = 0; nt < 4; ++nt) acc2[nt] = vzero8();
#pragma unroll
  for (int kt = 0; kt < 2; ++kt) {
    const int kb = kt * 32 + g * 8;
    v16h a = frag_from(&hbuf[wv][ln * HD + kb], &hbuf[wv][ln * HD + kb + 16]);
#pragma unroll
    for (int nt = 0; nt < 4; ++nt) {
      const _Float16* bp = cw2buf + ((kt * 4 + nt) << 9) + lane * 16;
      v16h b = frag_from(bp, bp + 8);
      acc2[nt] = wmma16(a, b, acc2[nt]);
    }
  }
#pragma unroll
  for (int nt = 0; nt < 4; ++nt) {
    float bias = b2[c * HD + nt * 16 + ln];
#pragma unroll
    for (int r = 0; r < 8; ++r) {
      float v = acc2[nt][r] + bias;
      v = v > 0.f ? v : 0.f;
      hbuf[wv][(r + 8 * g) * HD + nt * 16 + ln] = (_Float16)v;
    }
  }

  // ---- layer 3: M=16 N=16 (col 0 live) K=64 ----
  v8f acc3 = vzero8();
#pragma unroll
  for (int kt = 0; kt < 2; ++kt) {
    const int kb = kt * 32 + g * 8;
    v16h a = frag_from(&hbuf[wv][ln * HD + kb], &hbuf[wv][ln * HD + kb + 16]);
    const _Float16* bp = cw3buf + (kt << 9) + lane * 16;
    v16h b = frag_from(bp, bp + 8);
    acc3 = wmma16(a, b, acc3);
  }

  const float b3v = b3[c];
  if (ln == 0) {
#pragma unroll
    for (int r = 0; r < 8; ++r) cbf[wv][r + 8 * g] = acc3[r] + b3v;
  }

#pragma unroll
  for (int r = 0; r < 8; ++r) {
    int m = r + 8 * g;
    float cv = cbf[wv][m];
    float w = cv * 0.5f + 0.5f;
    w = w < 0.f ? 0.f : (w > 1.f ? 1.f : w);
    size_t idx = ((size_t)(mb + m) * NC + c) * ED + ln;
    float pv = pos[idx];
    float nv = neg[idx];
    emb[(size_t)(mb + m) * (ED * NC) + ln * NC + c] = pv * w + nv * (1.f - w);
    if (ln == 0) con[(size_t)(mb + m) * NC + c] = cv;
  }
}

// ---------- host ----------

extern "C" void kernel_launch(void* const* d_in, const int* in_sizes, int n_in,
                              void* d_out, int out_size, void* d_ws, size_t ws_size,
                              hipStream_t stream) {
  const float* x   = (const float*)d_in[0];
  const float* pW1 = (const float*)d_in[1];  const float* pb1 = (const float*)d_in[2];
  const float* pW2 = (const float*)d_in[3];  const float* pb2 = (const float*)d_in[4];
  const float* pW3 = (const float*)d_in[5];  const float* pb3 = (const float*)d_in[6];
  const float* nW1 = (const float*)d_in[7];  const float* nb1 = (const float*)d_in[8];
  const float* nW2 = (const float*)d_in[9];  const float* nb2 = (const float*)d_in[10];
  const float* nW3 = (const float*)d_in[11]; const float* nb3 = (const float*)d_in[12];
  const float* cW1 = (const float*)d_in[13]; const float* cb1 = (const float*)d_in[14];
  const float* cW2 = (const float*)d_in[15]; const float* cb2 = (const float*)d_in[16];
  const float* cW3 = (const float*)d_in[17]; const float* cb3 = (const float*)d_in[18];

  char* ws = (char*)d_ws;
  size_t off = 0;
  auto bump = [&](size_t bytes) { void* p = ws + off; off = (off + bytes + 255) & ~(size_t)255; return p; };

  _Float16* xh   = (_Float16*)bump((size_t)BB * INDIM * 2);
  _Float16* pw1h = (_Float16*)bump((size_t)NC * INDIM * HD * 2);
  _Float16* nw1h = (_Float16*)bump((size_t)NC * INDIM * HD * 2);
  _Float16* pw2h = (_Float16*)bump((size_t)NC * HD * HD * 2);
  _Float16* nw2h = (_Float16*)bump((size_t)NC * HD * HD * 2);
  _Float16* pw3h = (_Float16*)bump((size_t)NC * HD * ED * 2);
  _Float16* nw3h = (_Float16*)bump((size_t)NC * HD * ED * 2);
  _Float16* cw1h = (_Float16*)bump((size_t)NC * 32 * HD * 2);
  _Float16* cw2h = (_Float16*)bump((size_t)NC * HD * HD * 2);
  _Float16* cw3h = (_Float16*)bump((size_t)NC * HD * ED * 2);
  float* pos_out = (float*)bump((size_t)BB * NC * ED * 4);
  float* neg_out = (float*)bump((size_t)BB * NC * ED * 4);
  (void)ws_size; (void)n_in; (void)in_sizes; (void)out_size;

  // Phase 0
  {
    int n = BB * INDIM;
    cvt_f16_kernel<<<(n + 255) / 256, 256, 0, stream>>>(x, xh, n);
  }
  auto repack = [&](const float* W, _Float16* o, int Ktot, int Nsrc, int Ndst) {
    int total = NC * Ktot * Ndst;
    repack_w_kernel<<<(total + 255) / 256, 256, 0, stream>>>(W, o, Ktot, Nsrc, Ndst, total);
  };
  repack(pW1, pw1h, INDIM, HD, HD);
  repack(nW1, nw1h, INDIM, HD, HD);
  repack(pW2, pw2h, HD, HD, HD);
  repack(nW2, nw2h, HD, HD, HD);
  repack(pW3, pw3h, HD, ED, ED);
  repack(nW3, nw3h, HD, ED, ED);
  repack(cW1, cw1h, 32, HD, HD);
  repack(cW2, cw2h, HD, HD, HD);
  repack(cW3, cw3h, HD, 1, ED);

  // Phase 1: 8 waves/WG, 32 rows/wave -> 256 rows per WG
  dim3 gridB(BB / 256, NC), blk(256);
  bank_kernel<<<gridB, blk, 0, stream>>>(xh, pw1h, pb1, pw2h, pb2, pw3h, pb3, pos_out);
  bank_kernel<<<gridB, blk, 0, stream>>>(xh, nw1h, nb1, nw2h, nb2, nw3h, nb3, neg_out);

  // Phase 2
  float* emb = (float*)d_out;
  float* con = emb + (size_t)BB * ED * NC;
  dim3 gridC(BB / 128, NC);
  cp_gate_kernel<<<gridC, blk, 0, stream>>>(pos_out, neg_out, cw1h, cb1, cw2h, cb2,
                                            cw3h, cb3, emb, con);
}
